// MoeFc_85899345920455
// MI455X (gfx1250) — matrix operand order repaired and checked by reading, not compiled, not run on hardware
//
#include <hip/hip_runtime.h>
#include <math.h>

typedef __attribute__((ext_vector_type(16))) __bf16 vbf16;
typedef __attribute__((ext_vector_type(8)))  __bf16 vbf8;
typedef __attribute__((ext_vector_type(4)))  __bf16 vbf4;
typedef __attribute__((ext_vector_type(8)))  float  vf8;
typedef __attribute__((ext_vector_type(4)))  float  vf4;
typedef int vsi4 __attribute__((vector_size(16)));   // matches async builtin param type

#define TILE_M 64
#define TILE_N 128
#define TILE_K 32
#define LDSS   40   // LDS row stride in halves (80B: keeps 16B alignment, skews banks)

#if defined(__has_builtin)
#if __has_builtin(__builtin_amdgcn_global_load_async_to_lds_b128)
#define HAVE_ASYNC_LDS 1
#endif
#endif

#define AS1 __attribute__((address_space(1)))
#define AS3 __attribute__((address_space(3)))

static __device__ inline void wait_async0() {
#if defined(HAVE_ASYNC_LDS)
#if __has_builtin(__builtin_amdgcn_s_wait_asynccnt)
    __builtin_amdgcn_s_wait_asynccnt(0);
#else
    asm volatile("s_wait_asynccnt 0" ::: "memory");
#endif
#endif
}

static __device__ inline vbf16 combine8(vbf8 lo, vbf8 hi) {
    vbf16 r;
#pragma unroll
    for (int i = 0; i < 8; ++i) { r[i] = lo[i]; r[i + 8] = hi[i]; }
    return r;
}

// Stage one 64x32 A tile (async global->LDS) and one 32x128 B tile
// (f32 -> bf16, transposed to Bs[n][k], packed 4-wide along K -> ds_store_b64).
static __device__ inline void stage_tile(const __bf16* __restrict__ A,
                                         const float* __restrict__ W,
                                         int K, int N, int mBlk, int nBlk, int k0,
                                         __bf16* As, __bf16* Bs, int tid) {
#pragma unroll
    for (int i = 0; i < 2; ++i) {
        int idx = tid + i * 128;          // 0..255: 256 x 16B chunks
        int row = idx >> 2, kg = idx & 3; // 4 groups of 8 halves
        const __bf16* g = A + (size_t)(mBlk + row) * K + k0 + kg * 8;
        __bf16* l = &As[row * LDSS + kg * 8];
#if defined(HAVE_ASYNC_LDS)
        __builtin_amdgcn_global_load_async_to_lds_b128(
            (AS1 vsi4*)g, (AS3 vsi4*)l, 0, 0);
#else
        *reinterpret_cast<uint4*>(l) = *reinterpret_cast<const uint4*>(g);
#endif
    }
#pragma unroll
    for (int i = 0; i < 2; ++i) {
        int idx = tid + i * 128;          // 0..255
        int kq = idx >> 5, ng = idx & 31; // kq: group of 4 K rows; ng: group of 4 cols
        const float* base = W + (size_t)(k0 + kq * 4) * N + nBlk + ng * 4;
        vf4 r0 = *reinterpret_cast<const vf4*>(base);
        vf4 r1 = *reinterpret_cast<const vf4*>(base + N);
        vf4 r2 = *reinterpret_cast<const vf4*>(base + 2 * (size_t)N);
        vf4 r3 = *reinterpret_cast<const vf4*>(base + 3 * (size_t)N);
#pragma unroll
        for (int j = 0; j < 4; ++j) {
            vbf4 p;
            p[0] = (__bf16)r0[j]; p[1] = (__bf16)r1[j];
            p[2] = (__bf16)r2[j]; p[3] = (__bf16)r3[j];
            *reinterpret_cast<vbf4*>(&Bs[(ng * 4 + j) * LDSS + kq * 4]) = p;
        }
    }
}

// MODE 0: C = relu(A@W + bias) stored as bf16 (activation layer)
// MODE 1: v[e,:] += sum_t relu(A@W + bias)[t,:] * wtok[t,e]   (final layer fused reduce)
template <int MODE>
__global__ __launch_bounds__(128) void moe_gemm_kernel(
    const __bf16* __restrict__ A,   // [T x K] bf16 row-major
    const float*  __restrict__ W,   // [K x N] f32 row-major
    const float*  __restrict__ bias,// [N]
    __bf16*       __restrict__ Cout,// [T x N] bf16 (MODE 0)
    const float*  __restrict__ wtok,// [T x 8]      (MODE 1)
    int eidx,
    float*        __restrict__ vout,// [8 x N]      (MODE 1)
    int K, int N)
{
    __shared__ __align__(16) __bf16 As[2][TILE_M * LDSS];
    __shared__ __align__(16) __bf16 Bs[2][TILE_N * LDSS];
    __shared__ float vred[TILE_N];

    const int tid  = threadIdx.x;
    const int lane = tid & 31;
    const int wave = tid >> 5;
    const int wm = wave >> 1, wn = wave & 1;  // wave tile: rows 32*wm.., cols 64*wn..
    const int l16 = lane & 15, lh = lane >> 4;
    const int mBlk = blockIdx.x * TILE_M;
    const int nBlk = blockIdx.y * TILE_N;

    vf8 acc[2][4];
#pragma unroll
    for (int i = 0; i < 2; ++i)
#pragma unroll
        for (int j = 0; j < 4; ++j) acc[i][j] = (vf8)(0.0f);

    if (MODE == 1) vred[tid] = 0.0f;   // tid covers 0..127 == TILE_N-1

    const int NT = K / TILE_K;
    // Prologue: stage tile 0 into buffer 0
    stage_tile(A, W, K, N, mBlk, nBlk, 0, As[0], Bs[0], tid);

    for (int kt = 0; kt < NT; ++kt) {
        wait_async0();       // this wave's async copies for tile kt done
        __syncthreads();     // all waves' staging of tile kt visible
        const int cur = kt & 1;
        if (kt + 1 < NT)     // prefetch next tile into the other buffer
            stage_tile(A, W, K, N, mBlk, nBlk, (kt + 1) * TILE_K,
                       As[cur ^ 1], Bs[cur ^ 1], tid);

        const __bf16* Ac = As[cur];
        const __bf16* Bc = Bs[cur];
        // A fragments: lane half lh selects K 0-7/16-23 vs 8-15/24-31 (ISA layout)
        const int klo = lh * 8;
        vbf16 afrag[2], bfrag[4];
#pragma unroll
        for (int tm = 0; tm < 2; ++tm) {
            int M = 32 * wm + 16 * tm + l16;
            vbf8 lo = *reinterpret_cast<const vbf8*>(&Ac[M * LDSS + klo]);
            vbf8 hi = *reinterpret_cast<const vbf8*>(&Ac[M * LDSS + klo + 16]);
            afrag[tm] = combine8(lo, hi);
        }
        // B fragments: lane half selects K 0-7/16-23 vs 8-15/24-31 (same as A)
        const int kb = lh * 8;
#pragma unroll
        for (int tn = 0; tn < 4; ++tn) {
            int Nl = 64 * wn + 16 * tn + l16;
            vbf8 lo = *reinterpret_cast<const vbf8*>(&Bc[Nl * LDSS + kb]);
            vbf8 hi = *reinterpret_cast<const vbf8*>(&Bc[Nl * LDSS + kb + 16]);
            bfrag[tn] = combine8(lo, hi);
        }
#pragma unroll
        for (int tm = 0; tm < 2; ++tm)
#pragma unroll
            for (int tn = 0; tn < 4; ++tn)
                acc[tm][tn] = __builtin_amdgcn_wmma_f32_16x16x32_bf16(
                    false, afrag[tm], false, bfrag[tn], (short)0, acc[tm][tn],
                    false, false);
    }

    if (MODE == 0) {
#pragma unroll
        for (int tn = 0; tn < 4; ++tn) {
            int Ng = nBlk + 64 * wn + 16 * tn + l16;
            float bv = bias[Ng];
#pragma unroll
            for (int tm = 0; tm < 2; ++tm) {
                int Mb = mBlk + 32 * wm + 16 * tm + 8 * lh;
#pragma unroll
                for (int r = 0; r < 8; ++r) {
                    float f = acc[tm][tn][r] + bv;
                    f = fmaxf(f, 0.0f);
                    Cout[(size_t)(Mb + r) * N + Ng] = (__bf16)f;
                }
            }
        }
    } else {
        // token gate weights: independent of tn, load once
        float wt[2][8];
#pragma unroll
        for (int tm = 0; tm < 2; ++tm) {
            int Mb = mBlk + 32 * wm + 16 * tm + 8 * lh;
#pragma unroll
            for (int r = 0; r < 8; ++r)
                wt[tm][r] = wtok[(size_t)(Mb + r) * 8 + eidx];
        }
        float ps[4] = {0.0f, 0.0f, 0.0f, 0.0f};
#pragma unroll
        for (int tn = 0; tn < 4; ++tn) {
            int Ng = nBlk + 64 * wn + 16 * tn + l16;
            float bv = bias[Ng];
#pragma unroll
            for (int tm = 0; tm < 2; ++tm)
#pragma unroll
                for (int r = 0; r < 8; ++r) {
                    float f = fmaxf(acc[tm][tn][r] + bv, 0.0f);
                    ps[tn] += f * wt[tm][r];
                }
        }
        __syncthreads();
#pragma unroll
        for (int tn = 0; tn < 4; ++tn)
            atomicAdd(&vred[64 * wn + 16 * tn + l16], ps[tn]);
        __syncthreads();
        atomicAdd(&vout[(size_t)eidx * N + nBlk + tid], vred[tid]);
    }
}

// ---- gating pipeline -------------------------------------------------------

__global__ void cvt_bf16_kernel(const float* __restrict__ x,
                                __bf16* __restrict__ y, int n4) {
    int i = blockIdx.x * blockDim.x + threadIdx.x;
    if (i >= n4) return;
    float4 v = reinterpret_cast<const float4*>(x)[i];
    __bf16* o = y + (size_t)i * 4;
    o[0] = (__bf16)v.x; o[1] = (__bf16)v.y; o[2] = (__bf16)v.z; o[3] = (__bf16)v.w;
}

__global__ void gate_logits_kernel(const float* __restrict__ x,
                                   const float* __restrict__ gw,
                                   const float* __restrict__ gb,
                                   float* __restrict__ logits) {
    int idx = blockIdx.x * blockDim.x + threadIdx.x; // t*8 + e, 32768 total
    int t = idx >> 3, e = idx & 7;
    const float4* xr = reinterpret_cast<const float4*>(x + (size_t)t * 1024);
    float acc = gb[e];
    for (int d4 = 0; d4 < 256; ++d4) {
        float4 xv = xr[d4];
        int db = d4 * 4;
        acc += xv.x * gw[(db + 0) * 8 + e];
        acc += xv.y * gw[(db + 1) * 8 + e];
        acc += xv.z * gw[(db + 2) * 8 + e];
        acc += xv.w * gw[(db + 3) * 8 + e];
    }
    logits[idx] = acc;
}

// softmax over the sequence axis S (per (b,e) pair), S=1024
__global__ void softmax_s_kernel(const float* __restrict__ logits,
                                 float* __restrict__ probs) {
    __shared__ float red[256];
    int be = blockIdx.x;           // 0..31
    int b = be >> 3, e = be & 7;
    int tid = threadIdx.x;
    float l[4];
    float m = -__builtin_inff();
#pragma unroll
    for (int i = 0; i < 4; ++i) {
        int s = tid + i * 256;
        l[i] = logits[((size_t)(b << 10) + s) * 8 + e];
        m = fmaxf(m, l[i]);
    }
    red[tid] = m; __syncthreads();
    for (int off = 128; off > 0; off >>= 1) {
        if (tid < off) red[tid] = fmaxf(red[tid], red[tid + off]);
        __syncthreads();
    }
    float mx = red[0]; __syncthreads();
    float sum = 0.0f;
#pragma unroll
    for (int i = 0; i < 4; ++i) sum += expf(l[i] - mx);
    red[tid] = sum; __syncthreads();
    for (int off = 128; off > 0; off >>= 1) {
        if (tid < off) red[tid] += red[tid + off];
        __syncthreads();
    }
    float denom = red[0];
#pragma unroll
    for (int i = 0; i < 4; ++i) {
        int s = tid + i * 256;
        probs[((size_t)(b << 10) + s) * 8 + e] = expf(l[i] - mx) / denom;
    }
}

__global__ void topk_kernel(const float* __restrict__ probs,
                            float* __restrict__ wtok,
                            int* __restrict__ topidx) {
    int t = blockIdx.x * blockDim.x + threadIdx.x;
    if (t >= 4096) return;
    float p[8];
#pragma unroll
    for (int e = 0; e < 8; ++e) p[e] = probs[(size_t)t * 8 + e];
    int i1 = 0; float v1 = p[0];
#pragma unroll
    for (int e = 1; e < 8; ++e) if (p[e] > v1) { v1 = p[e]; i1 = e; }
    int i2 = -1; float v2 = -__builtin_inff();
#pragma unroll
    for (int e = 0; e < 8; ++e)
        if (e != i1 && p[e] > v2) { v2 = p[e]; i2 = e; }
#pragma unroll
    for (int e = 0; e < 8; ++e) wtok[(size_t)t * 8 + e] = 0.0f;
    wtok[(size_t)t * 8 + i1] = v1;
    wtok[(size_t)t * 8 + i2] = v2;
    topidx[t * 2]     = i1;
    topidx[t * 2 + 1] = i2;
}

// out[t,:] = v[e1,:] + v[e2,:]
__global__ void final_out_kernel(const int* __restrict__ topidx,
                                 const float* __restrict__ v,
                                 float* __restrict__ out) {
    int idx = blockIdx.x * blockDim.x + threadIdx.x; // t*256 + og (float4 groups)
    int t = idx >> 8, og = idx & 255;
    int i1 = topidx[t * 2], i2 = topidx[t * 2 + 1];
    float4 a = reinterpret_cast<const float4*>(v + (size_t)i1 * 1024)[og];
    float4 b = reinterpret_cast<const float4*>(v + (size_t)i2 * 1024)[og];
    float4 r = {a.x + b.x, a.y + b.y, a.z + b.z, a.w + b.w};
    reinterpret_cast<float4*>(out)[idx] = r;
}

extern "C" void kernel_launch(void* const* d_in, const int* in_sizes, int n_in,
                              void* d_out, int out_size, void* d_ws, size_t ws_size,
                              hipStream_t stream) {
    const float* x  = (const float*)d_in[0];
    const float* gw = (const float*)d_in[1];
    const float* gb = (const float*)d_in[2];
    const float* w1 = (const float*)d_in[3];
    const float* b1 = (const float*)d_in[4];
    const float* w2 = (const float*)d_in[5];
    const float* b2 = (const float*)d_in[6];
    const float* w3 = (const float*)d_in[7];
    const float* b3 = (const float*)d_in[8];
    float* out = (float*)d_out;
    char*  ws  = (char*)d_ws;

    const size_t T = 4096, D = 1024, O = 1024;

    __bf16* Xbf   = (__bf16*)(ws);                    // 8 MB
    __bf16* H1    = (__bf16*)(ws + (8u << 20));       // 8 MB
    __bf16* H2    = (__bf16*)(ws + (16u << 20));      // 8 MB
    float* logits = (float*)(ws + (24u << 20));                 // 128 KB
    float* probs  = (float*)(ws + (24u << 20) + (128u << 10));  // 128 KB
    float* wtok   = (float*)(ws + (24u << 20) + (256u << 10));  // 128 KB
    int*   topidx = (int*)  (ws + (24u << 20) + (384u << 10));  // 32 KB
    float* vbuf   = (float*)(ws + (24u << 20) + (416u << 10));  // 32 KB

    (void)hipMemsetAsync(vbuf, 0, 8 * O * sizeof(float), stream);

    cvt_bf16_kernel<<<(T * D / 4 + 255) / 256, 256, 0, stream>>>(x, Xbf, (int)(T * D / 4));
    gate_logits_kernel<<<(T * 8) / 256, 256, 0, stream>>>(x, gw, gb, logits);
    softmax_s_kernel<<<32, 256, 0, stream>>>(logits, probs);
    topk_kernel<<<16, 256, 0, stream>>>(probs, wtok, topidx);

    dim3 gg(T / TILE_M, O / TILE_N); // 64 x 8
    for (int e = 0; e < 8; ++e) {
        const size_t we = (size_t)e * D * O;
        moe_gemm_kernel<0><<<gg, 128, 0, stream>>>(Xbf, w1 + we, b1 + e * O, H1,
                                                   nullptr, 0, nullptr, (int)D, (int)O);
        moe_gemm_kernel<0><<<gg, 128, 0, stream>>>(H1, w2 + we, b2 + e * O, H2,
                                                   nullptr, 0, nullptr, (int)O, (int)O);
        moe_gemm_kernel<1><<<gg, 128, 0, stream>>>(H2, w3 + we, b3 + e * O, nullptr,
                                                   wtok, e, vbuf, (int)O, (int)O);
    }
    final_out_kernel<<<T, 256, 0, stream>>>(topidx, vbuf, out);
}
